// DecoderLSTM_15015205667069
// MI455X (gfx1250) — matrix operand order repaired and checked by reading, not compiled, not run on hardware
//
#include <hip/hip_runtime.h>
#include <hip/hip_bf16.h>

// ---------------------------------------------------------------------------
// Show-Attend-Tell decoder LSTM for MI455X (gfx1250).
// All GEMMs via v_wmma_f32_16x16x32_bf16 (fp32 accumulate). Weights are
// pre-swizzled once per call into WMMA-native per-lane layout so the B operand
// streams as aligned b128 loads; A loads are two aligned b128 per lane.
// Each wave computes two 16x16 N-tiles sharing one A fetch (2 WMMA / A load),
// and the k-loop is software-pipelined (double-buffered operands) so the next
// tile's VMEM flies under the current WMMAs instead of s_wait_loadcnt 0x0.
// W_fc bf16 (30.7MB) + all other weights are L2-resident (192MB L2); steady
// state is bound by the 146MB prediction writes (~6.3us HBM floor @ 23.3TB/s).
// ---------------------------------------------------------------------------

typedef __bf16 bf16_t;
typedef bf16_t v16bf __attribute__((ext_vector_type(16)));
typedef bf16_t v8bf  __attribute__((ext_vector_type(8)));
typedef float  v8f   __attribute__((ext_vector_type(8)));

#define B_   64
#define P_   196
#define E_   512
#define V_   30000
#define T_   20
#define NS_  19      // T-1 decode steps
#define XI_  1536    // [emb(512) | beta*att(512) | h(512)]

__device__ __forceinline__ float sigmoidf_(float x) { return 1.0f / (1.0f + __expf(-x)); }

// ---------------------------------------------------------------------------
// Stable descending argsort of 64 lengths via rank counting; also gathers the
// sorted captions and writes decode_len / sort_ind outputs.
// ---------------------------------------------------------------------------
__global__ void k_sort(const int* __restrict__ lens, const int* __restrict__ caps,
                       int* __restrict__ si_ws, int* __restrict__ dl_ws,
                       int* __restrict__ caps_out, int* __restrict__ dl_out,
                       int* __restrict__ si_out) {
  __shared__ int s_src[B_];
  int i = threadIdx.x;
  if (i < B_) {
    int li = lens[i];
    int rank = 0;
    for (int j = 0; j < B_; ++j) {
      int lj = lens[j];
      if (lj > li || (lj == li && j < i)) rank++;
    }
    si_ws[rank] = i;  si_out[rank] = i;
    dl_ws[rank] = li - 1;  dl_out[rank] = li - 1;
    s_src[rank] = i;
  }
  __syncthreads();
  for (int idx = i; idx < B_ * T_; idx += blockDim.x) {
    int b = idx / T_, t = idx % T_;
    caps_out[idx] = caps[s_src[b] * T_ + t];
  }
}

// ---------------------------------------------------------------------------
// fp32 weight (K x N row-major) -> bf16 in WMMA-native swizzled layout.
// Tile (nt, kt) occupies 512 contiguous bf16; within it lane L = half*16 + r
// owns 16 contiguous bf16: element j = W[kt*32 + j + half*16][nt*16 + r].
// A wave then loads its 16 B elements as two aligned b128 per k-step.
// ---------------------------------------------------------------------------
__global__ void k_f2b_sw(const float* __restrict__ W, bf16_t* __restrict__ dst,
                         int K, int N) {
  const long n = (long)K * N;
  const int KT = K >> 5;
  for (long i = (long)blockIdx.x * blockDim.x + threadIdx.x; i < n;
       i += (long)gridDim.x * blockDim.x) {
    int  j    = (int)(i & 15);
    int  lane = (int)((i >> 4) & 31);
    long t    = i >> 9;
    int  kt   = (int)(t % KT);
    int  nt   = (int)(t / KT);
    int  half = lane >> 4, r = lane & 15;
    int  kk   = kt * 32 + j + half * 16;
    int  col  = nt * 16 + r;
    dst[i] = (bf16_t)W[(long)kk * N + col];
  }
}

// Same, but reading the logical K-concatenation of two fp32 matrices
// (W_ih rows 0..K0-1, W_hh rows K0..K-1) -> one swizzled [W_ih;W_hh].
__global__ void k_f2b_sw2(const float* __restrict__ W0, int K0,
                          const float* __restrict__ W1,
                          bf16_t* __restrict__ dst, int K, int N) {
  const long n = (long)K * N;
  const int KT = K >> 5;
  for (long i = (long)blockIdx.x * blockDim.x + threadIdx.x; i < n;
       i += (long)gridDim.x * blockDim.x) {
    int  j    = (int)(i & 15);
    int  lane = (int)((i >> 4) & 31);
    long t    = i >> 9;
    int  kt   = (int)(t % KT);
    int  nt   = (int)(t / KT);
    int  half = lane >> 4, r = lane & 15;
    int  kk   = kt * 32 + j + half * 16;
    int  col  = nt * 16 + r;
    float v = (kk < K0) ? W0[(long)kk * N + col] : W1[(long)(kk - K0) * N + col];
    dst[i] = (bf16_t)v;
  }
}

// gather sorted encoder_out as bf16 (GEMM A operand for att_k/att_v)
__global__ void k_gather_eo(const float* __restrict__ eo, const int* __restrict__ si,
                            bf16_t* __restrict__ dst) {
  const long n = (long)B_ * P_ * E_;
  for (long i = (long)blockIdx.x * blockDim.x + threadIdx.x; i < n;
       i += (long)gridDim.x * blockDim.x) {
    int  e  = (int)(i % E_);
    long bp = i / E_;
    int  p  = (int)(bp % P_);
    int  b  = (int)(bp / P_);
    dst[i] = (bf16_t)eo[((long)si[b] * P_ + p) * E_ + e];
  }
}

// mean over P of sorted encoder_out -> bf16 (GEMM A operand for h0/c0)
__global__ void k_mean(const float* __restrict__ eo, const int* __restrict__ si,
                       bf16_t* __restrict__ mean_h) {
  int b = blockIdx.x;
  int src = si[b];
  for (int e = threadIdx.x; e < E_; e += blockDim.x) {
    float s = 0.f;
    const float* base = eo + (long)src * P_ * E_ + e;
    for (int p = 0; p < P_; ++p) s += base[(long)p * E_];
    mean_h[b * E_ + e] = (bf16_t)(s * (1.0f / (float)P_));
  }
}

// gather sorted caption embeddings (first NS_ steps) as bf16
__global__ void k_gather_emb(const float* __restrict__ emb, const int* __restrict__ caps_sorted,
                             bf16_t* __restrict__ dst) {
  const long n = (long)B_ * NS_ * E_;
  for (long i = (long)blockIdx.x * blockDim.x + threadIdx.x; i < n;
       i += (long)gridDim.x * blockDim.x) {
    int  e  = (int)(i % E_);
    long bt = i / E_;
    int  t  = (int)(bt % NS_);
    int  b  = (int)(bt / NS_);
    int tok = caps_sorted[b * T_ + t];
    dst[i] = (bf16_t)emb[(long)tok * E_ + e];
  }
}

// h0 fp32 -> bf16 into the h slice of xi
__global__ void k_h2xi(const float* __restrict__ h, bf16_t* __restrict__ xi) {
  int i = blockIdx.x * blockDim.x + threadIdx.x;
  if (i < B_ * E_) {
    int b = i / E_, e = i % E_;
    xi[(long)b * XI_ + 1024 + e] = (bf16_t)h[i];
  }
}

// ---------------------------------------------------------------------------
// A operand loader: contiguous k order matching the swizzled B (lanes 0-15
// K{0..15}, lanes 16-31 K{16..31}), loaded as two 16-byte runs per lane.
// ---------------------------------------------------------------------------
__device__ __forceinline__ v16bf load_a(const bf16_t* __restrict__ arow,
                                        int k0, int half) {
  v8bf lo = *(const v8bf*)(arow + k0 + half * 16);
  v8bf hi = *(const v8bf*)(arow + k0 + half * 16 + 8);
  v16bf a;
#pragma unroll
  for (int j = 0; j < 8; ++j) { a[j] = lo[j]; a[8 + j] = hi[j]; }
  return a;
}

// ---------------------------------------------------------------------------
// bf16 WMMA GEMM, B pre-swizzled. C[MxN] = A[MxK] * B + bias (+ row mask).
// 8 waves/block; each wave owns two adjacent 16x16 N-tiles (shared A fetch).
// K is a template constant (512 or 1536). The k-loop is software-pipelined:
// iteration kt+1's loads are issued before the WMMAs consuming iteration kt,
// so the backend can overlap VMEM with the matrix pipe (partial loadcnt waits).
// Prefetch is unconditional, locality 3 -> near-scope global_prefetch_b8.
// ---------------------------------------------------------------------------
template <int K, bool MASK>
__global__ void k_gemm(const bf16_t* __restrict__ A, int lda,
                       const bf16_t* __restrict__ Bsw,
                       const float* __restrict__ bias,
                       float* __restrict__ C, long ldc, int N,
                       const int* __restrict__ dl, int step) {
  constexpr int KT = K / 32;
  const int wave = threadIdx.x >> 5;
  const int lane = threadIdx.x & 31;
  const int half = lane >> 4;
  const int r    = lane & 15;
  const int NT   = N >> 4;
  const int nt0  = (blockIdx.x * (blockDim.x >> 5) + wave) * 2;
  const int m0   = blockIdx.y * 16;
  if (nt0 >= NT) return;                 // wave-uniform
  const bool two = (nt0 + 1) < NT;       // wave-uniform

  const bf16_t* arow = A + (long)(m0 + r) * lda;
  const bf16_t* pB0  = Bsw + (size_t)nt0 * KT * 512 + lane * 16;
  const bf16_t* pB1  = pB0 + (size_t)KT * 512;

  v8f acc0 = {}, acc1 = {};

  if (two) {
    v16bf a_c  = load_a(arow, 0, half);
    v16bf b0_c = *(const v16bf*)(pB0);
    v16bf b1_c = *(const v16bf*)(pB1);
#pragma unroll 2
    for (int kt = 0; kt < KT - 1; ++kt) {
      __builtin_prefetch(pB0 + (size_t)(kt + 4) * 512, 0, 3);
      __builtin_prefetch(pB1 + (size_t)(kt + 4) * 512, 0, 3);
      v16bf a_n  = load_a(arow, (kt + 1) * 32, half);
      v16bf b0_n = *(const v16bf*)(pB0 + (size_t)(kt + 1) * 512);
      v16bf b1_n = *(const v16bf*)(pB1 + (size_t)(kt + 1) * 512);
      acc0 = __builtin_amdgcn_wmma_f32_16x16x32_bf16(false, a_c, false, b0_c,
                                                     (short)0, acc0, false, false);
      acc1 = __builtin_amdgcn_wmma_f32_16x16x32_bf16(false, a_c, false, b1_c,
                                                     (short)0, acc1, false, false);
      a_c = a_n; b0_c = b0_n; b1_c = b1_n;
    }
    acc0 = __builtin_amdgcn_wmma_f32_16x16x32_bf16(false, a_c, false, b0_c,
                                                   (short)0, acc0, false, false);
    acc1 = __builtin_amdgcn_wmma_f32_16x16x32_bf16(false, a_c, false, b1_c,
                                                   (short)0, acc1, false, false);
  } else {
    v16bf a_c  = load_a(arow, 0, half);
    v16bf b0_c = *(const v16bf*)(pB0);
#pragma unroll 2
    for (int kt = 0; kt < KT - 1; ++kt) {
      __builtin_prefetch(pB0 + (size_t)(kt + 4) * 512, 0, 3);
      v16bf a_n  = load_a(arow, (kt + 1) * 32, half);
      v16bf b0_n = *(const v16bf*)(pB0 + (size_t)(kt + 1) * 512);
      acc0 = __builtin_amdgcn_wmma_f32_16x16x32_bf16(false, a_c, false, b0_c,
                                                     (short)0, acc0, false, false);
      a_c = a_n; b0_c = b0_n;
    }
    acc0 = __builtin_amdgcn_wmma_f32_16x16x32_bf16(false, a_c, false, b0_c,
                                                   (short)0, acc0, false, false);
  }

#pragma unroll
  for (int v = 0; v < 8; ++v) {
    int row = m0 + v + 8 * half;
    int col = nt0 * 16 + r;
    float out = acc0[v] + bias[col];
    if (MASK) out *= (step < dl[row]) ? 1.f : 0.f;   // row == batch here
    C[(long)row * ldc + col] = out;
  }
  if (two) {
#pragma unroll
    for (int v = 0; v < 8; ++v) {
      int row = m0 + v + 8 * half;
      int col = (nt0 + 1) * 16 + r;
      float out = acc1[v] + bias[col];
      if (MASK) out *= (step < dl[row]) ? 1.f : 0.f;
      C[(long)row * ldc + col] = out;
    }
  }
}

// ---------------------------------------------------------------------------
// scores = att_k . q * 1/sqrt(ATT); relu; softmax over P -> alpha (+ masked out)
// ---------------------------------------------------------------------------
__global__ void k_attention(const float* __restrict__ attk, const float* __restrict__ q,
                            const int* __restrict__ dl,
                            float* __restrict__ alpha, float* __restrict__ alphas_out,
                            int t) {
  int b = blockIdx.x;
  __shared__ float qs[E_];
  __shared__ float sc[P_];
  __shared__ float red[256];

  for (int e = threadIdx.x; e < E_; e += blockDim.x) qs[e] = q[b * E_ + e];
  __syncthreads();

  int p = threadIdx.x;
  if (p < P_) {
    const float* row = attk + ((long)b * P_ + p) * E_;
    float s = 0.f;
    for (int e = 0; e < E_; ++e) s += row[e] * qs[e];
    s *= 0.044194173824159216f;  // 1/sqrt(512)
    sc[p] = fmaxf(s, 0.f);       // relu
  }
  __syncthreads();

  float m = -1e30f;
  for (int i = threadIdx.x; i < P_; i += blockDim.x) m = fmaxf(m, sc[i]);
  red[threadIdx.x] = m;
  __syncthreads();
  for (int s2 = 128; s2 > 0; s2 >>= 1) {
    if (threadIdx.x < s2) red[threadIdx.x] = fmaxf(red[threadIdx.x], red[threadIdx.x + s2]);
    __syncthreads();
  }
  m = red[0];
  __syncthreads();

  float sum = 0.f;
  for (int i = threadIdx.x; i < P_; i += blockDim.x) {
    float e2 = __expf(sc[i] - m);
    sc[i] = e2;
    sum += e2;
  }
  red[threadIdx.x] = sum;
  __syncthreads();
  for (int s2 = 128; s2 > 0; s2 >>= 1) {
    if (threadIdx.x < s2) red[threadIdx.x] += red[threadIdx.x + s2];
    __syncthreads();
  }
  float inv = 1.f / red[0];
  float mf  = (t < dl[b]) ? 1.f : 0.f;
  if (p < P_) {
    float a = sc[p] * inv;
    alpha[b * P_ + p] = a;
    alphas_out[((long)b * NS_ + t) * P_ + p] = a * mf;
  }
}

// ---------------------------------------------------------------------------
// att_out = sigmoid(h@W_beta+b) * (alpha @ att_v); assemble bf16 xi = [emb|att|h]
// ---------------------------------------------------------------------------
__global__ void k_assemble(const float* __restrict__ attv, const float* __restrict__ alpha,
                           const float* __restrict__ betapre, const bf16_t* __restrict__ embs,
                           bf16_t* __restrict__ xi, int t) {
  int b = blockIdx.x;
  __shared__ float al[P_];
  for (int i = threadIdx.x; i < P_; i += blockDim.x) al[i] = alpha[b * P_ + i];
  __syncthreads();
  for (int e = threadIdx.x; e < E_; e += blockDim.x) {
    float s = 0.f;
    const float* col = attv + (long)b * P_ * E_ + e;
    for (int p = 0; p < P_; ++p) s += al[p] * col[(long)p * E_];
    float bet = sigmoidf_(betapre[b * E_ + e]);
    xi[(long)b * XI_ + 512 + e] = (bf16_t)(bet * s);
    xi[(long)b * XI_ + e]       = embs[((long)b * NS_ + t) * E_ + e];
  }
}

// ---------------------------------------------------------------------------
// LSTM cell: gates hold xi@[W_ih;W_hh]+b_ih; add b_hh here. Writes masked h/c
// (fp32), masked h bf16 into xi, and unmasked h_new bf16 for the pred GEMM.
// ---------------------------------------------------------------------------
__global__ void k_cell(const float* __restrict__ gates, const float* __restrict__ b_hh,
                       const int* __restrict__ dl,
                       float* __restrict__ h, float* __restrict__ c,
                       bf16_t* __restrict__ xi, bf16_t* __restrict__ hnew_h, int t) {
  int i = blockIdx.x * blockDim.x + threadIdx.x;
  if (i >= B_ * E_) return;
  int b = i / E_, e = i % E_;
  const float* g = gates + (long)b * 2048;
  float ig = g[e]          + b_hh[e];
  float fg = g[512 + e]    + b_hh[512 + e];
  float gg = g[1024 + e]   + b_hh[1024 + e];
  float og = g[1536 + e]   + b_hh[1536 + e];
  float cn = sigmoidf_(fg) * c[i] + sigmoidf_(ig) * tanhf(gg);
  float hn = sigmoidf_(og) * tanhf(cn);
  bool  m  = (t < dl[b]);
  float hm = m ? hn : h[i];
  float cm = m ? cn : c[i];
  h[i] = hm;
  c[i] = cm;
  xi[(long)b * XI_ + 1024 + e] = (bf16_t)hm;
  hnew_h[i] = (bf16_t)hn;
}

// ---------------------------------------------------------------------------
extern "C" void kernel_launch(void* const* d_in, const int* in_sizes, int n_in,
                              void* d_out, int out_size, void* d_ws, size_t ws_size,
                              hipStream_t stream) {
  (void)in_sizes; (void)n_in; (void)out_size; (void)ws_size;

  const float* eo     = (const float*)d_in[0];
  const int*   caps   = (const int*)d_in[1];
  const int*   lens   = (const int*)d_in[2];
  const float* Wk     = (const float*)d_in[3];
  const float* bk     = (const float*)d_in[4];
  const float* Wq     = (const float*)d_in[5];
  const float* bq     = (const float*)d_in[6];
  const float* Wv     = (const float*)d_in[7];
  const float* bv     = (const float*)d_in[8];
  const float* emb    = (const float*)d_in[9];
  const float* Wih    = (const float*)d_in[10];
  const float* bih    = (const float*)d_in[11];
  const float* Whh    = (const float*)d_in[12];
  const float* bhh    = (const float*)d_in[13];
  const float* Winih  = (const float*)d_in[14];
  const float* binih  = (const float*)d_in[15];
  const float* Winic  = (const float*)d_in[16];
  const float* binic  = (const float*)d_in[17];
  const float* Wbeta  = (const float*)d_in[18];
  const float* bbeta  = (const float*)d_in[19];
  const float* Wfc    = (const float*)d_in[20];
  const float* bfc    = (const float*)d_in[21];

  // d_out: preds (B,NS,V) f32 | alphas (B,NS,P) f32 | caps (B,T) i32 | dl (B) | si (B)
  float* preds_out  = (float*)d_out;
  float* alphas_out = preds_out + (size_t)B_ * NS_ * V_;
  int*   caps_out   = (int*)(alphas_out + (size_t)B_ * NS_ * P_);
  int*   dl_out     = caps_out + B_ * T_;
  int*   si_out     = dl_out + B_;

  // workspace carve-out
  char* w = (char*)d_ws;
  size_t off = 0;
  auto take = [&](size_t bytes) -> char* {
    char* p = w + off;
    off = (off + bytes + 255) & ~(size_t)255;
    return p;
  };
  int*    si_ws   = (int*)take(B_ * 4);
  int*    dl_ws   = (int*)take(B_ * 4);
  bf16_t* Wk_h    = (bf16_t*)take((size_t)E_ * E_ * 2);      // swizzled bf16
  bf16_t* Wq_h    = (bf16_t*)take((size_t)E_ * E_ * 2);
  bf16_t* Wv_h    = (bf16_t*)take((size_t)E_ * E_ * 2);
  bf16_t* Wbeta_h = (bf16_t*)take((size_t)E_ * E_ * 2);
  bf16_t* Winih_h = (bf16_t*)take((size_t)E_ * E_ * 2);
  bf16_t* Winic_h = (bf16_t*)take((size_t)E_ * E_ * 2);
  bf16_t* Wcat_h  = (bf16_t*)take((size_t)XI_ * 2048 * 2);   // [W_ih; W_hh]
  bf16_t* Wfc_h   = (bf16_t*)take((size_t)E_ * V_ * 2);
  bf16_t* eo_s_h  = (bf16_t*)take((size_t)B_ * P_ * E_ * 2);
  bf16_t* mean_h  = (bf16_t*)take((size_t)B_ * E_ * 2);
  bf16_t* embs_h  = (bf16_t*)take((size_t)B_ * NS_ * E_ * 2);
  bf16_t* xi_h    = (bf16_t*)take((size_t)B_ * XI_ * 2);
  bf16_t* hnew_h  = (bf16_t*)take((size_t)B_ * E_ * 2);
  float*  attk_f  = (float*)take((size_t)B_ * P_ * E_ * 4);
  float*  attv_f  = (float*)take((size_t)B_ * P_ * E_ * 4);
  float*  h_f     = (float*)take((size_t)B_ * E_ * 4);
  float*  c_f     = (float*)take((size_t)B_ * E_ * 4);
  float*  q_f     = (float*)take((size_t)B_ * E_ * 4);
  float*  beta_f  = (float*)take((size_t)B_ * E_ * 4);
  float*  alpha_f = (float*)take((size_t)B_ * P_ * 4);
  float*  gates_f = (float*)take((size_t)B_ * 2048 * 4);

  auto swz = [&](const float* s, bf16_t* d, int K, int N) {
    long n = (long)K * N;
    int blocks = (int)((n + 255) / 256);
    if (blocks > 1024) blocks = 1024;
    k_f2b_sw<<<blocks, 256, 0, stream>>>(s, d, K, N);
  };
  auto gemm512 = [&](const bf16_t* A, int lda, const bf16_t* Bsw, const float* bias,
                     float* C, long ldc, int M, int N, const int* dl, int step) {
    int NT = N / 16, pairs = (NT + 1) / 2;
    dim3 grid((pairs + 7) / 8, M / 16);
    if (dl)
      k_gemm<512, true><<<grid, 256, 0, stream>>>(A, lda, Bsw, bias, C, ldc, N, dl, step);
    else
      k_gemm<512, false><<<grid, 256, 0, stream>>>(A, lda, Bsw, bias, C, ldc, N,
                                                   nullptr, 0);
  };

  // ---- preamble ----
  k_sort<<<1, 64, 0, stream>>>(lens, caps, si_ws, dl_ws, caps_out, dl_out, si_out);

  swz(Wk, Wk_h, E_, E_);
  swz(Wq, Wq_h, E_, E_);
  swz(Wv, Wv_h, E_, E_);
  swz(Wbeta, Wbeta_h, E_, E_);
  swz(Winih, Winih_h, E_, E_);
  swz(Winic, Winic_h, E_, E_);
  swz(Wfc, Wfc_h, E_, V_);
  k_f2b_sw2<<<1024, 256, 0, stream>>>(Wih, 1024, Whh, Wcat_h, XI_, 2048);

  k_gather_eo<<<1024, 256, 0, stream>>>(eo, si_ws, eo_s_h);
  k_mean<<<B_, 256, 0, stream>>>(eo, si_ws, mean_h);
  k_gather_emb<<<1024, 256, 0, stream>>>(emb, caps_out, embs_h);

  // h0 / c0
  gemm512(mean_h, E_, Winih_h, binih, h_f, E_, B_, E_, nullptr, 0);
  gemm512(mean_h, E_, Winic_h, binic, c_f, E_, B_, E_, nullptr, 0);
  k_h2xi<<<(B_ * E_ + 255) / 256, 256, 0, stream>>>(h_f, xi_h);

  // att_k / att_v over all (b,p) rows at once: M = B*P = 12544
  gemm512(eo_s_h, E_, Wk_h, bk, attk_f, E_, B_ * P_, E_, nullptr, 0);
  gemm512(eo_s_h, E_, Wv_h, bv, attv_f, E_, B_ * P_, E_, nullptr, 0);

  // ---- sequential decode steps ----
  for (int t = 0; t < NS_; ++t) {
    // q = h @ Wq + bq ; beta_pre = h @ W_beta + b_beta  (h slice of xi, lda=XI_)
    gemm512(xi_h + 1024, XI_, Wq_h, bq, q_f, E_, B_, E_, nullptr, 0);
    gemm512(xi_h + 1024, XI_, Wbeta_h, bbeta, beta_f, E_, B_, E_, nullptr, 0);

    k_attention<<<B_, 256, 0, stream>>>(attk_f, q_f, dl_ws, alpha_f, alphas_out, t);
    k_assemble<<<B_, 256, 0, stream>>>(attv_f, alpha_f, beta_f, embs_h, xi_h, t);

    // gates = [emb | b*att | h] @ [W_ih; W_hh] + b_ih   (b_hh added in cell)
    {
      int NT = 2048 / 16, pairs = (NT + 1) / 2;
      dim3 grid((pairs + 7) / 8, B_ / 16);
      k_gemm<1536, false><<<grid, 256, 0, stream>>>(xi_h, XI_, Wcat_h, bih,
                                                    gates_f, 2048, 2048, nullptr, 0);
    }

    k_cell<<<(B_ * E_ + 255) / 256, 256, 0, stream>>>(gates_f, bhh, dl_ws, h_f, c_f,
                                                      xi_h, hnew_h, t);

    // pred[:, t, :] = (h_new @ W_fc + b_fc) * mask  (written straight into d_out)
    gemm512(hnew_h, E_, Wfc_h, bfc, preds_out + (long)t * V_, (long)NS_ * V_,
            B_, V_, dl_ws, t);
  }
}
